// MambaSubBlock_9328668967272
// MI455X (gfx1250) — matrix operand order repaired and checked by reading, not compiled, not run on hardware
//
#include <hip/hip_runtime.h>
#include <hip/hip_bf16.h>
#include <math.h>

// ---------------- problem constants ----------------
constexpr int BB   = 8;
constexpr int LL   = 4096;     // 64*64
constexpr int CD   = 128;      // DIM
constexpr int DI   = 256;      // D_INNER
constexpr int DSN  = 16;       // D_STATE
constexpr int XND  = 40;       // DT_RANK + 2*D_STATE (logical)
constexpr int XNDP = 64;       // padded x_dbl row stride
constexpr int HIDN = 512;
constexpr size_t NTOK = (size_t)BB * LL;   // 32768

// workspace layout (in floats)
constexpr size_t O_TOK  = 0;
constexpr size_t O_NORM = O_TOK  + NTOK * CD;
constexpr size_t O_OH   = O_NORM + NTOK * CD;
constexpr size_t O_OV   = O_OH   + NTOK * CD;
constexpr size_t O_XZ   = O_OV   + NTOK * CD;     // NTOK*512 (reused as MLP hidden)
constexpr size_t O_XC   = O_XZ   + NTOK * 512;    // NTOK*256
constexpr size_t O_XDBL = O_XC   + NTOK * 256;    // NTOK*64 (padded)
constexpr size_t O_Y    = O_XDBL + NTOK * XNDP;   // NTOK*256
constexpr size_t O_CTX  = O_Y    + NTOK * 256;    // BB*CD
constexpr size_t O_GATE = O_CTX  + (size_t)BB * CD;
constexpr size_t O_WXP  = O_GATE + (size_t)BB * CD;   // 64*256 padded Wx

// ---------------- WMMA types ----------------
typedef __attribute__((ext_vector_type(16))) __bf16 v16bf;
typedef __attribute__((ext_vector_type(8)))  float  v8f;

// sequence-position -> token-index map for the 4 directional passes
// 0: h forward   1: h reversed   2: v forward   3: v reversed
__device__ __forceinline__ int rowmapf(int s, int mode) {
  switch (mode) {
    case 0: return s;
    case 1: return (LL - 1) - s;
    case 2: return ((s & 63) << 6) | (s >> 6);
    default: { int s2 = (LL - 1) - s; return ((s2 & 63) << 6) | (s2 >> 6); }
  }
}

// pack 16 gathered floats into a bf16x16 fragment
__device__ __forceinline__ v16bf pack16(const float4& a0, const float4& a1,
                                        const float4& a2, const float4& a3) {
  v16bf v;
  v[0]=(__bf16)a0.x;  v[1]=(__bf16)a0.y;  v[2]=(__bf16)a0.z;  v[3]=(__bf16)a0.w;
  v[4]=(__bf16)a1.x;  v[5]=(__bf16)a1.y;  v[6]=(__bf16)a1.z;  v[7]=(__bf16)a1.w;
  v[8]=(__bf16)a2.x;  v[9]=(__bf16)a2.y;  v[10]=(__bf16)a2.z; v[11]=(__bf16)a2.w;
  v[12]=(__bf16)a3.x; v[13]=(__bf16)a3.y; v[14]=(__bf16)a3.z; v[15]=(__bf16)a3.w;
  return v;
}

__device__ __forceinline__ v16bf load_frag(const float* p) {
  float4 a0 = *(const float4*)(p + 0);
  float4 a1 = *(const float4*)(p + 4);
  float4 a2 = *(const float4*)(p + 8);
  float4 a3 = *(const float4*)(p + 12);
  return pack16(a0, a1, a2, a3);
}

// ---------------- WMMA GEMM: C[M,N] = act(A[M,K] x W[N,K]^T + bias) ----------------
// Divergence-free: N must be a multiple of 16 and covered by full 64-col wave
// strips (host guarantees N in {64,128,512}). One shared A fragment feeds 4
// independent WMMA accumulator chains per K-step. A rows gathered via mapA,
// C rows scattered via mapC, optional +=. Requires M == BB*LL, K % 32 == 0.
__global__ void
gemm_wmma_kernel(const float* __restrict__ A, const float* __restrict__ W,
                 const float* __restrict__ bias, float* __restrict__ C,
                 int N, int K, int mapA, int mapC, int accum, int act)
{
  const int wave = threadIdx.x >> 5;
  const int lane = threadIdx.x & 31;
  const int half = lane >> 4;
  const int mn   = lane & 15;

  const int wavesPerBlock = blockDim.x >> 5;
  const int mtile = blockIdx.x << 4;
  const int n0w   = (blockIdx.y * wavesPerBlock + wave) << 6;  // 64-col strip
  if (n0w >= N) return;

  // A row for this lane (lanes 0-15 / 16-31 both cover M=0..15, half = K-phase)
  const int rlin  = mtile + mn;
  const int bA    = rlin >> 12;
  const int sA    = rlin & (LL - 1);
  // A frag: K = j + 16*half
  const float* pa = A + ((size_t)(bA << 12) + (size_t)rowmapf(sA, mapA)) * (size_t)K
                      + (half << 4);
  // B frag: lane = column n, K = j + 16*half
  const size_t wstep = (size_t)K;
  const float* pw0 = W + (size_t)(n0w      + mn) * wstep + (half << 4);
  const float* pw1 = W + (size_t)(n0w + 16 + mn) * wstep + (half << 4);
  const float* pw2 = W + (size_t)(n0w + 32 + mn) * wstep + (half << 4);
  const float* pw3 = W + (size_t)(n0w + 48 + mn) * wstep + (half << 4);

  v8f acc0 = {}, acc1 = {}, acc2 = {}, acc3 = {};

  for (int kk = 0; kk < K; kk += 32) {
    if (kk + 32 < K) __builtin_prefetch(pa + kk + 32, 0, 1);   // global_prefetch_b8
    // A fragment: 16 contiguous floats
    float4 a0 = *(const float4*)(pa + kk + 0);
    float4 a1 = *(const float4*)(pa + kk + 4);
    float4 a2 = *(const float4*)(pa + kk + 8);
    float4 a3 = *(const float4*)(pa + kk + 12);
    const v16bf av = pack16(a0, a1, a2, a3);

    const v16bf bv0 = load_frag(pw0 + kk);
    const v16bf bv1 = load_frag(pw1 + kk);
    const v16bf bv2 = load_frag(pw2 + kk);
    const v16bf bv3 = load_frag(pw3 + kk);

    acc0 = __builtin_amdgcn_wmma_f32_16x16x32_bf16(false, av, false, bv0, (short)0, acc0, false, false);
    acc1 = __builtin_amdgcn_wmma_f32_16x16x32_bf16(false, av, false, bv1, (short)0, acc1, false, false);
    acc2 = __builtin_amdgcn_wmma_f32_16x16x32_bf16(false, av, false, bv2, (short)0, acc2, false, false);
    acc3 = __builtin_amdgcn_wmma_f32_16x16x32_bf16(false, av, false, bv3, (short)0, acc3, false, false);
  }

  // ---- epilogue: bias / activation / (scatter-)store, full tiles, no guards
  #pragma unroll
  for (int r8 = 0; r8 < 8; ++r8) {
    const int rr  = mtile + r8 + (half << 3);   // C VGPR r holds M = r + 8*half
    const int bC  = rr >> 12;
    const int sC  = rr & (LL - 1);
    const size_t crow = ((size_t)(bC << 12) + (size_t)rowmapf(sC, mapC)) * (size_t)N;
    float v0 = acc0[r8], v1 = acc1[r8], v2 = acc2[r8], v3 = acc3[r8];
    if (bias) {
      v0 += bias[n0w + mn];      v1 += bias[n0w + 16 + mn];
      v2 += bias[n0w + 32 + mn]; v3 += bias[n0w + 48 + mn];
    }
    if (act == 1) {
      v0 = 0.5f * v0 * (1.0f + erff(v0 * 0.70710678118654752f));
      v1 = 0.5f * v1 * (1.0f + erff(v1 * 0.70710678118654752f));
      v2 = 0.5f * v2 * (1.0f + erff(v2 * 0.70710678118654752f));
      v3 = 0.5f * v3 * (1.0f + erff(v3 * 0.70710678118654752f));
    }
    float* cr = C + crow + n0w + mn;
    if (accum) {
      cr[0] += v0; cr[16] += v1; cr[32] += v2; cr[48] += v3;
    } else {
      cr[0] = v0; cr[16] = v1; cr[32] = v2; cr[48] = v3;
    }
  }
}

// ---------------- pad Wx (40x256) into zero-filled (64x256) ----------------
__global__ void pad_wx_kernel(const float* __restrict__ Wx, float* __restrict__ WxP)
{
  const int i = blockIdx.x * blockDim.x + threadIdx.x;   // over 64*256
  const int n = i >> 8, k = i & (DI - 1);
  WxP[i] = (n < XND) ? Wx[n * DI + k] : 0.0f;
}

// ---------------- depthwise causal conv (k=4) + SiLU ----------------
__global__ void conv_silu_kernel(const float* __restrict__ xz,
                                 const float* __restrict__ cw,
                                 const float* __restrict__ cb,
                                 float* __restrict__ xc)
{
  const int idx = blockIdx.x * blockDim.x + threadIdx.x;   // over NTOK*256
  const int d = idx & (DI - 1);
  const int t = idx >> 8;
  const int s = t & (LL - 1);
  float acc = cb[d];
  #pragma unroll
  for (int k = 0; k < 4; ++k) {
    const int sp = s - 3 + k;
    if (sp >= 0) acc += cw[d * 4 + k] * xz[(size_t)(t - 3 + k) * 512 + d];
  }
  const float sg = 1.0f / (1.0f + __expf(-acc));
  xc[(size_t)t * DI + d] = acc * sg;
}

// ---------------- selective-scan: 1 block per batch, thread d owns 16 states ----------------
__global__ void __launch_bounds__(256)
scan_kernel(const float* __restrict__ xdbl, const float* __restrict__ xc,
            const float* __restrict__ xz,   const float* __restrict__ Wdt,
            const float* __restrict__ bdt,  const float* __restrict__ Alog,
            const float* __restrict__ Dp,   float* __restrict__ y)
{
  const int b = blockIdx.x;
  const int d = threadIdx.x;

  float Avec[DSN];
  #pragma unroll
  for (int nn = 0; nn < DSN; ++nn) Avec[nn] = -__expf(Alog[d * DSN + nn]);
  float wdt[8];
  #pragma unroll
  for (int j = 0; j < 8; ++j) wdt[j] = Wdt[d * 8 + j];
  const float bd = bdt[d];
  const float Dd = Dp[d];

  float h[DSN];
  #pragma unroll
  for (int nn = 0; nn < DSN; ++nn) h[nn] = 0.0f;

  __shared__ float sB[2][DSN];
  __shared__ float sC[2][DSN];

  for (int s = 0; s < LL; ++s) {
    const size_t base = ((size_t)b << 12) + (size_t)s;
    const int buf = s & 1;
    if (d < 32) {
      const float v = xdbl[base * XNDP + 8 + d];   // cols 8..23 = B, 24..39 = C
      if (d < 16) sB[buf][d] = v; else sC[buf][d - 16] = v;
    }
    const float* di = xdbl + base * XNDP;
    float dt = bd;
    #pragma unroll
    for (int j = 0; j < 8; ++j) dt += wdt[j] * di[j];
    dt = (dt > 20.0f) ? dt : __logf(1.0f + __expf(dt));       // softplus
    const float u = xc[base * DI + d];
    const float z = xz[base * 512 + DI + d];
    __syncthreads();
    float yv = 0.0f;
    const float dtu = dt * u;
    #pragma unroll
    for (int nn = 0; nn < DSN; ++nn) {
      h[nn] = h[nn] * __expf(dt * Avec[nn]) + dtu * sB[buf][nn];
      yv += h[nn] * sC[buf][nn];
    }
    const float sg = 1.0f / (1.0f + __expf(-z));
    y[base * DI + d] = (yv + u * Dd) * (z * sg);
  }
}

// ---------------- initial NCHW->tokens + LayerNorm1 ----------------
__global__ void __launch_bounds__(128)
tok_ln_kernel(const float* __restrict__ x, const float* __restrict__ g,
              const float* __restrict__ bta, float* __restrict__ tok,
              float* __restrict__ norm)
{
  const int t = blockIdx.x, c = threadIdx.x;
  const int b = t >> 12, l = t & (LL - 1);
  const float xv = x[((size_t)b * CD + c) * (size_t)LL + l];
  const size_t idx = (size_t)t * CD + c;
  tok[idx] = xv;
  __shared__ float red[CD];
  red[c] = xv; __syncthreads();
  for (int st = 64; st > 0; st >>= 1) { if (c < st) red[c] += red[c + st]; __syncthreads(); }
  const float mu = red[0] * (1.0f / CD); __syncthreads();
  const float dv = xv - mu;
  red[c] = dv * dv; __syncthreads();
  for (int st = 64; st > 0; st >>= 1) { if (c < st) red[c] += red[c + st]; __syncthreads(); }
  const float var = red[0] * (1.0f / CD);
  norm[idx] = dv * rsqrtf(var + 1e-5f) * g[c] + bta[c];
}

// ---------------- context mean: 1 block per (b,c) ----------------
__global__ void __launch_bounds__(256)
ctx_kernel(const float* __restrict__ oh, const float* __restrict__ ov,
           float* __restrict__ ctx)
{
  const int bc = blockIdx.x;
  const int b = bc >> 7, c = bc & (CD - 1);
  const float* po = oh + ((size_t)b << 12) * CD + c;
  const float* pv = ov + ((size_t)b << 12) * CD + c;
  float sum = 0.0f;
  for (int l = threadIdx.x; l < LL; l += 256)
    sum += 0.5f * (po[(size_t)l * CD] + pv[(size_t)l * CD]);
  __shared__ float red[256];
  red[threadIdx.x] = sum; __syncthreads();
  for (int st = 128; st > 0; st >>= 1) { if (threadIdx.x < st) red[threadIdx.x] += red[threadIdx.x + st]; __syncthreads(); }
  if (threadIdx.x == 0) ctx[bc] = red[0] * (1.0f / LL);
}

// ---------------- gate MLP (tiny): 1 block per batch ----------------
__global__ void __launch_bounds__(128)
gate_kernel(const float* __restrict__ ctx, const float* __restrict__ W1,
            const float* __restrict__ b1, const float* __restrict__ W2,
            const float* __restrict__ b2, float* __restrict__ gate)
{
  const int b = blockIdx.x, c = threadIdx.x;
  __shared__ float sc[CD], hid[32];
  sc[c] = ctx[b * CD + c]; __syncthreads();
  if (c < 32) {
    float a = b1[c];
    for (int j = 0; j < CD; ++j) a += W1[c * CD + j] * sc[j];
    hid[c] = a > 0.0f ? a : 0.0f;
  }
  __syncthreads();
  float a = b2[c];
  for (int j = 0; j < 32; ++j) a += W2[c * 32 + j] * hid[j];
  gate[b * CD + c] = 1.0f / (1.0f + __expf(-a));
}

// ---------------- fuse + residual + LayerNorm2 ----------------
__global__ void __launch_bounds__(128)
fuse_ln_kernel(const float* __restrict__ oh, const float* __restrict__ ov,
               const float* __restrict__ gate, const float* __restrict__ g2,
               const float* __restrict__ b2, float* __restrict__ tok,
               float* __restrict__ norm)
{
  const int t = blockIdx.x, c = threadIdx.x;
  const int b = t >> 12;
  const size_t idx = (size_t)t * CD + c;
  const float g = gate[b * CD + c];
  const float f = g * oh[idx] + (1.0f - g) * ov[idx];
  const float xv = tok[idx] + f;
  tok[idx] = xv;
  __shared__ float red[CD];
  red[c] = xv; __syncthreads();
  for (int st = 64; st > 0; st >>= 1) { if (c < st) red[c] += red[c + st]; __syncthreads(); }
  const float mu = red[0] * (1.0f / CD); __syncthreads();
  const float dv = xv - mu;
  red[c] = dv * dv; __syncthreads();
  for (int st = 64; st > 0; st >>= 1) { if (c < st) red[c] += red[c + st]; __syncthreads(); }
  const float var = red[0] * (1.0f / CD);
  norm[idx] = dv * rsqrtf(var + 1e-5f) * g2[c] + b2[c];
}

// ---------------- tokens -> NCHW output ----------------
__global__ void __launch_bounds__(128)
untranspose_kernel(const float* __restrict__ tok, float* __restrict__ out)
{
  const int t = blockIdx.x, c = threadIdx.x;
  const int b = t >> 12, l = t & (LL - 1);
  out[((size_t)b * CD + c) * (size_t)LL + l] = tok[(size_t)t * CD + c];
}

// ---------------- host orchestration ----------------
extern "C" void kernel_launch(void* const* d_in, const int* in_sizes, int n_in,
                              void* d_out, int out_size, void* d_ws, size_t ws_size,
                              hipStream_t stream)
{
  (void)in_sizes; (void)n_in; (void)out_size; (void)ws_size;
  const float* x    = (const float*)d_in[0];
  const float* n1_g = (const float*)d_in[1];
  const float* n1_b = (const float*)d_in[2];
  const float* gW1  = (const float*)d_in[21];
  const float* gb1  = (const float*)d_in[22];
  const float* gW2  = (const float*)d_in[23];
  const float* gb2  = (const float*)d_in[24];
  const float* n2_g = (const float*)d_in[25];
  const float* n2_b = (const float*)d_in[26];
  const float* mW1  = (const float*)d_in[27];
  const float* mb1  = (const float*)d_in[28];
  const float* mW2  = (const float*)d_in[29];
  const float* mb2  = (const float*)d_in[30];

  float* ws   = (float*)d_ws;
  float* tok  = ws + O_TOK;
  float* norm = ws + O_NORM;
  float* oh   = ws + O_OH;
  float* ov   = ws + O_OV;
  float* xz   = ws + O_XZ;      // also reused as MLP hidden
  float* xc   = ws + O_XC;
  float* xdbl = ws + O_XDBL;    // ld = 64, cols 0..39 valid
  float* ybuf = ws + O_Y;
  float* ctx  = ws + O_CTX;
  float* gate = ws + O_GATE;
  float* WxP  = ws + O_WXP;     // 64x256 zero-padded Wx

  const dim3 blk32(32), blk64(64), blk128(128), blk256(256);
  const int MT = (int)(NTOK / 16);   // 2048 M-tiles

  // 1) NCHW -> tokens, LayerNorm1
  tok_ln_kernel<<<(int)NTOK, blk128, 0, stream>>>(x, n1_g, n1_b, tok, norm);

  // 2) zero directional accumulators
  hipMemsetAsync(oh, 0, NTOK * CD * sizeof(float), stream);
  hipMemsetAsync(ov, 0, NTOK * CD * sizeof(float), stream);

  // 3) four directional Mamba passes
  for (int pass = 0; pass < 4; ++pass) {
    const int base = (pass < 2) ? 3 : 12;
    const int mode = pass;                 // 0:h-fwd 1:h-rev 2:v-fwd 3:v-rev
    float* accb = (pass < 2) ? oh : ov;
    const float* Win  = (const float*)d_in[base + 0];
    const float* cw   = (const float*)d_in[base + 1];
    const float* cb   = (const float*)d_in[base + 2];
    const float* Wx   = (const float*)d_in[base + 3];
    const float* Wdt  = (const float*)d_in[base + 4];
    const float* bdt  = (const float*)d_in[base + 5];
    const float* Alog = (const float*)d_in[base + 6];
    const float* Dp   = (const float*)d_in[base + 7];
    const float* Wout = (const float*)d_in[base + 8];

    // zero-pad Wx to 64 rows (removes all guards from the GEMM)
    pad_wx_kernel<<<(64 * DI) / 256, blk256, 0, stream>>>(Wx, WxP);

    // xz = gather(norm) @ Win^T   (M=32768, N=512, K=128): 4 waves, grid.y=2
    gemm_wmma_kernel<<<dim3(MT, 2), blk128, 0, stream>>>(
        norm, Win, nullptr, xz, 512, CD, mode, 0, 0, 0);

    // depthwise causal conv + SiLU on x-half of xz
    conv_silu_kernel<<<(int)(NTOK * DI / 256), blk256, 0, stream>>>(xz, cw, cb, xc);

    // x_dbl = xc @ WxP^T   (N=64 padded, K=256): 1 wave per block
    gemm_wmma_kernel<<<dim3(MT, 1), blk32, 0, stream>>>(
        xc, WxP, nullptr, xdbl, XNDP, DI, 0, 0, 0, 0);

    // selective scan (dt fused) -> y
    scan_kernel<<<BB, blk256, 0, stream>>>(xdbl, xc, xz, Wdt, bdt, Alog, Dp, ybuf);

    // acc[map] += y @ Wout^T   (N=128, K=256): 2 waves per block
    gemm_wmma_kernel<<<dim3(MT, 1), blk64, 0, stream>>>(
        ybuf, Wout, nullptr, accb, CD, DI, 0, mode, 1, 0);
  }

  // 4) context + gate
  ctx_kernel<<<BB * CD, blk256, 0, stream>>>(oh, ov, ctx);
  gate_kernel<<<BB, blk128, 0, stream>>>(ctx, gW1, gb1, gW2, gb2, gate);

  // 5) fuse + residual + LayerNorm2
  fuse_ln_kernel<<<(int)NTOK, blk128, 0, stream>>>(oh, ov, gate, n2_g, n2_b, tok, norm);

  // 6) MLP: hid = GELU(norm @ W1^T + b1)   (reuse xz buffer)
  gemm_wmma_kernel<<<dim3(MT, 2), blk128, 0, stream>>>(
      norm, mW1, mb1, xz, HIDN, CD, 0, 0, 0, 1);
  //    tok += hid @ W2^T + b2
  gemm_wmma_kernel<<<dim3(MT, 1), blk64, 0, stream>>>(
      xz, mW2, mb2, tok, CD, HIDN, 0, 0, 1, 0);

  // 7) tokens -> NCHW output
  untranspose_kernel<<<(int)NTOK, blk128, 0, stream>>>(tok, (float*)d_out);
}
